// SimGNN_66537633349987
// MI455X (gfx1250) — matrix operand order repaired and checked by reading, not compiled, not run on hardware
//
#include <hip/hip_runtime.h>
#include <hip/hip_bf16.h>
#include <math.h>

// ---------------- problem constants ----------------
#define NN   32768      // total nodes (B*NPG)
#define DD   128        // hidden dim
#define EE   524288     // edges per edge set
#define BB   64         // graphs
#define NPGc 512        // nodes per graph
#define KK   16         // NTN slices
#define BINSc 16

typedef __attribute__((ext_vector_type(16))) _Float16 v16h;
typedef __attribute__((ext_vector_type(8)))  float    v8f;
typedef __attribute__((ext_vector_type(4)))  int      v4i;

union V16U { v16h v; unsigned u[8]; };

// ---------------- async global->LDS staging (CDNA5 ASYNCcnt path) ----------------
#if __has_builtin(__builtin_amdgcn_global_load_async_to_lds_b128) && \
    __has_builtin(__builtin_amdgcn_s_wait_asynccnt)
#define USE_ASYNC_LDS 1
typedef __attribute__((address_space(1))) v4i* gvec_p;
typedef __attribute__((address_space(3))) v4i* lvec_p;
#else
#define USE_ASYNC_LDS 0
#endif

__device__ __forceinline__ void cp16_to_lds(const _Float16* g, _Float16* s) {
#if USE_ASYNC_LDS
  __builtin_amdgcn_global_load_async_to_lds_b128(
      (gvec_p)(unsigned long long)(uintptr_t)g,
      (lvec_p)(unsigned long long)(uintptr_t)s,
      0, 0);
#else
  *reinterpret_cast<float4*>(s) = *reinterpret_cast<const float4*>(g);
#endif
}

__device__ __forceinline__ void stage_fence_sync() {
#if USE_ASYNC_LDS
  __builtin_amdgcn_s_wait_asynccnt(0);
#endif
  __syncthreads();
}

// ---- WMMA operand loads per CDNA5 ISA 7.12.2 (16-bit A 16x32 / B 32x16) ----
// A: lane L -> row m=L&15, half h=L>>4; vgpr v holds K = kt*32 + (v>>2)*16 + h*8 + (v&3)*2 (+0,+1)
__device__ __forceinline__ v16h load_a16(const _Float16* __restrict__ tile, int kt, int lane) {
  int m = lane & 15, h = lane >> 4;
  V16U r;
  const _Float16* row = tile + m * DD;
  #pragma unroll
  for (int v = 0; v < 8; ++v) {
    int k = kt * 32 + ((v >> 2) << 4) + (h << 3) + ((v & 3) << 1);
    r.u[v] = *reinterpret_cast<const unsigned*>(row + k);
  }
  return r.v;
}
// B[k][n] = Bt[n*128 + k] (Bt row-major [outCol, K]); lane L -> col n=L&15, half h=L>>4;
// vgpr v holds K = kt*32 + (v>>2)*16 + h*8 + (v&3)*2 (+0,+1)
__device__ __forceinline__ v16h load_b16(const _Float16* tile, int kt, int lane) {
  int n = lane & 15, h = lane >> 4;
  V16U r;
  const _Float16* row = tile + n * DD;
  #pragma unroll
  for (int v = 0; v < 8; ++v) {
    int k = kt * 32 + ((v >> 2) << 4) + (h << 3) + ((v & 3) << 1);
    r.u[v] = *reinterpret_cast<const unsigned*>(row + k);
  }
  return r.v;
}

// ---------------- generic K=128 GEMM: C[M,Ncols] = A[M,128] * Bt[Ncols,128]^T ----------------
// grid: x = ceil(rowTiles/8), y = colStrips. Block = 256 thr = 8 waves sharing one
// 128-col B strip staged in LDS (32KB). EPI 0: plain store. EPI 1: tanh(v + ebias[col]).
template<int EPI>
__global__ void gemm128_kernel(const _Float16* __restrict__ A, const _Float16* __restrict__ Bt,
                               float* __restrict__ C, int rowTiles, int colStrips,
                               const float* __restrict__ ebias) {
  __shared__ __align__(16) _Float16 Bs[128 * DD];
  int t  = threadIdx.x;
  int cs = blockIdx.y;
  const _Float16* Bg = Bt + (size_t)cs * 128 * DD;
  #pragma unroll
  for (int i = 0; i < 8; ++i)
    cp16_to_lds(Bg + (size_t)(i * 256 + t) * 8, Bs + (size_t)(i * 256 + t) * 8);
  stage_fence_sync();

  int gw   = (int)blockIdx.x * 8 + (t >> 5);
  int lane = t & 31;
  if (gw >= rowTiles) return;
  const _Float16* Ab = A + (size_t)gw * 16 * DD;

  v8f acc[8];
  #pragma unroll
  for (int i = 0; i < 8; ++i) acc[i] = v8f{0.f,0.f,0.f,0.f,0.f,0.f,0.f,0.f};

  #pragma unroll
  for (int kt = 0; kt < 4; ++kt) {
    v16h a = load_a16(Ab, kt, lane);
    __builtin_prefetch(Ab + (lane & 15) * DD + ((kt + 1) & 3) * 32, 0, 1);
    #pragma unroll
    for (int ct = 0; ct < 8; ++ct) {
      v16h b = load_b16(Bs + (size_t)ct * 16 * DD, kt, lane);   // LDS
      acc[ct] = __builtin_amdgcn_wmma_f32_16x16x32_f16(false, a, false, b, (short)0,
                                                       acc[ct], false, false);
    }
  }

  int n = lane & 15, h = lane >> 4;
  size_t Ncols = (size_t)colStrips * 128;
  #pragma unroll
  for (int ct = 0; ct < 8; ++ct) {
    int col = cs * 128 + ct * 16 + n;
    #pragma unroll
    for (int r = 0; r < 8; ++r) {
      int row = gw * 16 + h * 8 + r;
      float v = acc[ct][r];
      if (EPI == 1) v = tanhf(v + ebias[col]);
      C[(size_t)row * Ncols + col] = v;
    }
  }
}

// ---------------- fused cosine-sim GEMM + histogram binning ----------------
// 128 blocks/graph: block = (mt, ntg) with A row-block (16x128, 4KB) and B col-block
// (128x128, 32KB) staged in LDS; 8 waves each compute one 16x16 sims tile and bin it.
__global__ void hist_kernel(const _Float16* __restrict__ n1f, const _Float16* __restrict__ n2f,
                            float* __restrict__ hist) {
  __shared__ __align__(16) _Float16 As[16 * DD];
  __shared__ __align__(16) _Float16 Bs[128 * DD];
  __shared__ unsigned sh[BINSc];
  int t = threadIdx.x;
  if (t < BINSc) sh[t] = 0u;

  int b   = (int)(blockIdx.x >> 7);
  int rem = (int)(blockIdx.x & 127);
  int mt = rem >> 2, ntg = rem & 3;
  const _Float16* Ag = n1f + (size_t)(b * NPGc + mt * 16) * DD;
  const _Float16* Bg = n2f + (size_t)(b * NPGc + ntg * 128) * DD;

  cp16_to_lds(Ag + (size_t)t * 8, As + (size_t)t * 8);
  #pragma unroll
  for (int i = 0; i < 8; ++i)
    cp16_to_lds(Bg + (size_t)(i * 256 + t) * 8, Bs + (size_t)(i * 256 + t) * 8);
  stage_fence_sync();

  int wave = t >> 5, lane = t & 31;
  const _Float16* Bb = Bs + (size_t)wave * 16 * DD;

  v8f acc = v8f{0.f,0.f,0.f,0.f,0.f,0.f,0.f,0.f};
  #pragma unroll
  for (int kt = 0; kt < 4; ++kt) {
    v16h a  = load_a16(As, kt, lane);
    v16h bb = load_b16(Bb, kt, lane);
    acc = __builtin_amdgcn_wmma_f32_16x16x32_f16(false, a, false, bb, (short)0, acc, false, false);
  }
  #pragma unroll
  for (int r = 0; r < 8; ++r) {
    float s = acc[r];
    int bi = (int)floorf((s + 1.0f) * 8.0f);
    bi = bi < 0 ? 0 : (bi > 15 ? 15 : bi);
    atomicAdd(&sh[bi], 1u);                         // LDS ds_add_u32
  }
  __syncthreads();
  if (t < BINSc) atomicAdd(&hist[b * BINSc + t], (float)sh[t]);
}

// ---------------- small helper kernels ----------------
__global__ void zero_kernel(float* __restrict__ p, int n) {
  int i = blockIdx.x * 256 + threadIdx.x;
  if (i < n) p[i] = 0.f;
}

__global__ void cvt_f16_kernel(const float* __restrict__ x, _Float16* __restrict__ y, int n) {
  int i = blockIdx.x * 256 + threadIdx.x;
  if (i < n) y[i] = (_Float16)x[i];
}

// ntn_T [K,D,D] (k,d,e) -> Tt [(k*128+e), d] f16 (so B[k_dim=d][col=k*128+e] pairs are contiguous)
__global__ void ttn_kernel(const float* __restrict__ T, _Float16* __restrict__ Tt) {
  int i = blockIdx.x * 256 + threadIdx.x;
  if (i >= KK * DD * DD) return;
  int k = i >> 14, d = (i >> 7) & 127, e = i & 127;
  Tt[(size_t)(k * DD + e) * DD + d] = (_Float16)T[i];
}

// degree with self loops (deg buffer must be zeroed)
__global__ void deg_kernel(const int* __restrict__ ei, float* __restrict__ deg) {
  int i = blockIdx.x * 256 + threadIdx.x;
  if (i < EE)            atomicAdd(&deg[ei[EE + i]], 1.f);   // dst = ei[1][e]
  else if (i < EE + NN)  atomicAdd(&deg[i - EE], 1.f);       // self loop
}

__global__ void rsqrt_kernel(float* __restrict__ d) {
  int i = blockIdx.x * 256 + threadIdx.x;
  if (i < NN) { float v = d[i]; d[i] = (v > 0.f) ? rsqrtf(v) : 0.f; }
}

// out[dst] += dinv[src]*dinv[dst]*h[src]  (E real edges + N self loops); 4 channels/thread
__global__ void aggregate_kernel(const int* __restrict__ ei, const float* __restrict__ dinv,
                                 const float* __restrict__ h, float* __restrict__ out) {
  long long tid  = (long long)blockIdx.x * blockDim.x + threadIdx.x;
  long long item = tid >> 5;
  if (item >= (long long)(EE + NN)) return;
  int c = ((int)tid & 31) << 2;
  int src, dst;
  if (item < EE) { src = ei[item]; dst = ei[EE + item]; }
  else           { src = dst = (int)(item - EE); }
  float nrm = dinv[src] * dinv[dst];
  const float4 hv = *reinterpret_cast<const float4*>(h + (size_t)src * DD + c);
  float* o = out + (size_t)dst * DD + c;
  atomicAdd(o + 0, nrm * hv.x);
  atomicAdd(o + 1, nrm * hv.y);
  atomicAdd(o + 2, nrm * hv.z);
  atomicAdd(o + 3, nrm * hv.w);
}

// x = relu(BN(x + gcn_b)); writes f32 (in place) and f16 copy for next GEMM
__global__ void bn_relu_kernel(float* __restrict__ x, _Float16* __restrict__ xf,
                               const float* __restrict__ gb,  const float* __restrict__ gam,
                               const float* __restrict__ bet, const float* __restrict__ mean,
                               const float* __restrict__ var, int l) {
  int i = blockIdx.x * 256 + threadIdx.x;
  if (i >= NN * DD) return;
  int c = i & 127;
  int lc = l * DD + c;
  float v = x[i] + gb[lc];
  v = (v - mean[lc]) * rsqrtf(var[lc] + 1e-5f) * gam[lc] + bet[lc];
  v = fmaxf(v, 0.f);
  x[i]  = v;
  xf[i] = (_Float16)v;
}

// s[n] = dot(t[n,:], att_w) + att_b  (one wave per node)
__global__ void score_kernel(const float* __restrict__ t, const float* __restrict__ att_w,
                             const float* __restrict__ att_b, float* __restrict__ s) {
  int wv   = (int)((blockIdx.x * blockDim.x + threadIdx.x) >> 5);
  int lane = threadIdx.x & 31;
  if (wv >= NN) return;
  const float* row = t + (size_t)wv * DD;
  float acc = 0.f;
  #pragma unroll
  for (int j = 0; j < 4; ++j) acc += row[lane + 32 * j] * att_w[lane + 32 * j];
  #pragma unroll
  for (int off = 16; off; off >>= 1) acc += __shfl_xor(acc, off);
  if (lane == 0) s[wv] = acc + att_b[0];
}

// per-graph softmax over 512 scores, then weighted sum of embeddings -> g[b,128]
__global__ void pool_kernel(const float* __restrict__ s, const float* __restrict__ emb,
                            float* __restrict__ g) {
  __shared__ float red[256];
  __shared__ float w[NPGc];
  int b = blockIdx.x, t = threadIdx.x;
  const float* sb = s + b * NPGc;
  float m = -1e30f;
  for (int i = t; i < NPGc; i += 256) m = fmaxf(m, sb[i]);
  red[t] = m; __syncthreads();
  for (int off = 128; off; off >>= 1) { if (t < off) red[t] = fmaxf(red[t], red[t + off]); __syncthreads(); }
  float mx = red[0]; __syncthreads();
  float sum = 0.f;
  for (int i = t; i < NPGc; i += 256) { float e = expf(sb[i] - mx); w[i] = e; sum += e; }
  red[t] = sum; __syncthreads();
  for (int off = 128; off; off >>= 1) { if (t < off) red[t] += red[t + off]; __syncthreads(); }
  float inv = 1.f / red[0];
  if (t < DD) {
    float acc = 0.f;
    const float* eb = emb + (size_t)b * NPGc * DD + t;
    for (int nrow = 0; nrow < NPGc; ++nrow) acc += w[nrow] * eb[(size_t)nrow * DD];
    g[b * DD + t] = acc * inv;
  }
}

// row-normalize emb (f32) -> f16 (overwrites the activation f16 buffer)
__global__ void rownorm_kernel(const float* __restrict__ e, _Float16* __restrict__ nf) {
  int wv   = (int)((blockIdx.x * blockDim.x + threadIdx.x) >> 5);
  int lane = threadIdx.x & 31;
  if (wv >= NN) return;
  const float* row = e + (size_t)wv * DD;
  float v[4]; float ss = 0.f;
  #pragma unroll
  for (int j = 0; j < 4; ++j) { v[j] = row[lane + 32 * j]; ss += v[j] * v[j]; }
  #pragma unroll
  for (int off = 16; off; off >>= 1) ss += __shfl_xor(ss, off);
  float inv = 1.f / fmaxf(sqrtf(ss), 1e-8f);
  _Float16* orow = nf + (size_t)wv * DD;
  #pragma unroll
  for (int j = 0; j < 4; ++j) orow[lane + 32 * j] = (_Float16)(v[j] * inv);
}

// tp[b,k] = dot(M1[b, k*128 : k*128+128], g2[b,:]) + ntn_bias[k]
__global__ void tp_kernel(const float* __restrict__ M1, const float* __restrict__ g2,
                          const float* __restrict__ tb, float* __restrict__ tp) {
  int b = blockIdx.x, wv = threadIdx.x >> 5, lane = threadIdx.x & 31;
  for (int kk = wv; kk < KK; kk += 8) {
    const float* row = M1 + (size_t)b * (KK * DD) + kk * DD;
    const float* gg  = g2 + (size_t)b * DD;
    float acc = 0.f;
    #pragma unroll
    for (int j = 0; j < 4; ++j) acc += row[lane + 32 * j] * gg[lane + 32 * j];
    #pragma unroll
    for (int off = 16; off; off >>= 1) acc += __shfl_xor(acc, off);
    if (lane == 0) tp[b * KK + kk] = acc + tb[kk];
  }
}

__device__ __forceinline__ float sigmf(float x) { return 1.f / (1.f + expf(-x)); }

// hist normalization + hist MLP + NTN MLP + fusion MLP; one thread per graph
__global__ void final_kernel(const float* __restrict__ tp, const float* __restrict__ hist,
                             const float* __restrict__ nw1, const float* __restrict__ nb1,
                             const float* __restrict__ nw2, const float* __restrict__ nb2,
                             const float* __restrict__ nw3, const float* __restrict__ nb3,
                             const float* __restrict__ hw1, const float* __restrict__ hb1,
                             const float* __restrict__ hw2, const float* __restrict__ hb2,
                             const float* __restrict__ fw1, const float* __restrict__ fb1,
                             const float* __restrict__ fw2, const float* __restrict__ fb2,
                             float* __restrict__ out) {
  int b = threadIdx.x;
  if (b >= BB) return;
  float hn[BINSc]; float hs = 0.f;
  for (int i = 0; i < BINSc; ++i) hs += hist[b * BINSc + i];
  hs += 1e-8f;
  for (int i = 0; i < BINSc; ++i) hn[i] = hist[b * BINSc + i] / hs;
  float h1[32], h2[16];
  for (int o = 0; o < 32; ++o) {
    float a = hb1[o];
    for (int i = 0; i < 16; ++i) a += hn[i] * hw1[o * 16 + i];
    h1[o] = fmaxf(a, 0.f);
  }
  for (int o = 0; o < 16; ++o) {
    float a = hb2[o];
    for (int i = 0; i < 32; ++i) a += h1[i] * hw2[o * 32 + i];
    h2[o] = a;
  }
  float m1[32], m2[16];
  for (int o = 0; o < 32; ++o) {
    float a = nb1[o];
    for (int i = 0; i < 16; ++i) a += tp[b * KK + i] * nw1[o * 16 + i];
    m1[o] = fmaxf(a, 0.f);
  }
  for (int o = 0; o < 16; ++o) {
    float a = nb2[o];
    for (int i = 0; i < 32; ++i) a += m1[i] * nw2[o * 32 + i];
    m2[o] = fmaxf(a, 0.f);
  }
  float ns = nb3[0];
  for (int i = 0; i < 16; ++i) ns += m2[i] * nw3[i];
  ns = sigmf(ns);
  float comb[17];
  comb[0] = ns;
  for (int i = 0; i < 16; ++i) comb[1 + i] = h2[i];
  float f1[8];
  for (int o = 0; o < 8; ++o) {
    float a = fb1[o];
    for (int i = 0; i < 17; ++i) a += comb[i] * fw1[o * 17 + i];
    f1[o] = fmaxf(a, 0.f);
  }
  float y = fb2[0];
  for (int i = 0; i < 8; ++i) y += f1[i] * fw2[i];
  out[b] = sigmf(y);
}

// ---------------- launch ----------------
extern "C" void kernel_launch(void* const* d_in, const int* in_sizes, int n_in,
                              void* d_out, int out_size, void* d_ws, size_t ws_size,
                              hipStream_t stream) {
  (void)in_sizes; (void)n_in; (void)out_size; (void)ws_size;

  const float* x1    = (const float*)d_in[0];
  const float* x2    = (const float*)d_in[1];
  const int*   ei1   = (const int*)d_in[2];
  const int*   ei2   = (const int*)d_in[3];
  const float* gcn_w = (const float*)d_in[6];
  const float* gcn_b = (const float*)d_in[7];
  const float* bn_g  = (const float*)d_in[8];
  const float* bn_b  = (const float*)d_in[9];
  const float* bn_m  = (const float*)d_in[10];
  const float* bn_v  = (const float*)d_in[11];
  const float* ctx_w = (const float*)d_in[12];
  const float* ctx_b = (const float*)d_in[13];
  const float* att_w = (const float*)d_in[14];
  const float* att_b = (const float*)d_in[15];
  const float* ntn_T = (const float*)d_in[16];
  const float* ntn_bias = (const float*)d_in[17];
  const float* ntn_w1 = (const float*)d_in[18];
  const float* ntn_b1 = (const float*)d_in[19];
  const float* ntn_w2 = (const float*)d_in[20];
  const float* ntn_b2 = (const float*)d_in[21];
  const float* ntn_w3 = (const float*)d_in[22];
  const float* ntn_b3 = (const float*)d_in[23];
  const float* hist_w1 = (const float*)d_in[24];
  const float* hist_b1 = (const float*)d_in[25];
  const float* hist_w2 = (const float*)d_in[26];
  const float* hist_b2 = (const float*)d_in[27];
  const float* fus_w1 = (const float*)d_in[28];
  const float* fus_b1 = (const float*)d_in[29];
  const float* fus_w2 = (const float*)d_in[30];
  const float* fus_b2 = (const float*)d_in[31];

  char* ws = (char*)d_ws;
  size_t off = 0;
  auto take = [&](size_t bytes) -> char* {
    off = (off + 255) & ~(size_t)255;
    char* p = ws + off;
    off += bytes;
    return p;
  };

  float*     emb1   = (float*)take((size_t)NN * DD * 4);
  float*     emb2   = (float*)take((size_t)NN * DD * 4);
  float*     hbuf   = (float*)take((size_t)NN * DD * 4);
  _Float16*  xf1    = (_Float16*)take((size_t)NN * DD * 2);
  _Float16*  xf2    = (_Float16*)take((size_t)NN * DD * 2);
  float*     dinv1  = (float*)take((size_t)NN * 4);
  float*     dinv2  = (float*)take((size_t)NN * 4);
  _Float16*  gcnw16 = (_Float16*)take((size_t)3 * DD * DD * 2);
  _Float16*  ctxw16 = (_Float16*)take((size_t)DD * DD * 2);
  _Float16*  Tt16   = (_Float16*)take((size_t)KK * DD * DD * 2);
  float*     sbuf   = (float*)take((size_t)NN * 4);
  float*     g1     = (float*)take((size_t)BB * DD * 4);
  float*     g2     = (float*)take((size_t)BB * DD * 4);
  _Float16*  g1f    = (_Float16*)take((size_t)BB * DD * 2);
  _Float16*  g2f    = (_Float16*)take((size_t)BB * DD * 2);
  float*     M1     = (float*)take((size_t)BB * KK * DD * 4);
  float*     tpb    = (float*)take((size_t)BB * KK * 4);
  float*     histb  = (float*)take((size_t)BB * BINSc * 4);

  auto blocks = [](long long n, int bs) -> unsigned { return (unsigned)((n + bs - 1) / bs); };

  // weight conversions
  cvt_f16_kernel<<<blocks(3 * DD * DD, 256), 256, 0, stream>>>(gcn_w, gcnw16, 3 * DD * DD);
  cvt_f16_kernel<<<blocks(DD * DD, 256), 256, 0, stream>>>(ctx_w, ctxw16, DD * DD);
  ttn_kernel<<<blocks(KK * DD * DD, 256), 256, 0, stream>>>(ntn_T, Tt16);

  // degree + sym-norm
  zero_kernel<<<blocks(NN, 256), 256, 0, stream>>>(dinv1, NN);
  zero_kernel<<<blocks(NN, 256), 256, 0, stream>>>(dinv2, NN);
  deg_kernel<<<blocks(EE + NN, 256), 256, 0, stream>>>(ei1, dinv1);
  deg_kernel<<<blocks(EE + NN, 256), 256, 0, stream>>>(ei2, dinv2);
  rsqrt_kernel<<<blocks(NN, 256), 256, 0, stream>>>(dinv1);
  rsqrt_kernel<<<blocks(NN, 256), 256, 0, stream>>>(dinv2);

  const float*   xs[2]  = { x1, x2 };
  const int*     eis[2] = { ei1, ei2 };
  float*         dvs[2] = { dinv1, dinv2 };
  float*         ems[2] = { emb1, emb2 };
  _Float16*      xfs[2] = { xf1, xf2 };
  float*         gs[2]  = { g1, g2 };

  const dim3 gemmNgrid(NN / 16 / 8, 1);     // 2048 row tiles, 8 waves/block, 1 col strip
  for (int s = 0; s < 2; ++s) {
    cvt_f16_kernel<<<blocks((long long)NN * DD, 256), 256, 0, stream>>>(xs[s], xfs[s], NN * DD);
    for (int l = 0; l < 3; ++l) {
      gemm128_kernel<0><<<gemmNgrid, 256, 0, stream>>>(xfs[s], gcnw16 + (size_t)l * DD * DD,
                                                       hbuf, NN / 16, 1, nullptr);
      zero_kernel<<<blocks((long long)NN * DD, 256), 256, 0, stream>>>(ems[s], NN * DD);
      aggregate_kernel<<<blocks((long long)(EE + NN) * 32, 256), 256, 0, stream>>>(
          eis[s], dvs[s], hbuf, ems[s]);
      bn_relu_kernel<<<blocks((long long)NN * DD, 256), 256, 0, stream>>>(
          ems[s], xfs[s], gcn_b, bn_g, bn_b, bn_m, bn_v, l);
    }
    // attention pooling
    gemm128_kernel<1><<<gemmNgrid, 256, 0, stream>>>(xfs[s], ctxw16, hbuf, NN / 16, 1, ctx_b);
    score_kernel<<<blocks((long long)NN * 32, 256), 256, 0, stream>>>(hbuf, att_w, att_b, sbuf);
    pool_kernel<<<BB, 256, 0, stream>>>(sbuf, ems[s], gs[s]);
    // normalized f16 embeddings for histogram (overwrites activation f16 buffer)
    rownorm_kernel<<<blocks((long long)NN * 32, 256), 256, 0, stream>>>(ems[s], xfs[s]);
  }

  // fused cosine-sim GEMM + histogram
  zero_kernel<<<blocks(BB * BINSc, 256), 256, 0, stream>>>(histb, BB * BINSc);
  hist_kernel<<<BB * 128, 256, 0, stream>>>(xf1, xf2, histb);

  // NTN bilinear: M1 = g1 @ T  ([64,128] x [128, 16*128]), then tp = <M1, g2>
  cvt_f16_kernel<<<blocks(BB * DD, 256), 256, 0, stream>>>(g1, g1f, BB * DD);
  cvt_f16_kernel<<<blocks(BB * DD, 256), 256, 0, stream>>>(g2, g2f, BB * DD);
  gemm128_kernel<0><<<dim3(1, KK), 256, 0, stream>>>(g1f, Tt16, M1, BB / 16, KK, nullptr);
  tp_kernel<<<BB, 256, 0, stream>>>(M1, g2, ntn_bias, tpb);

  // heads + fusion
  final_kernel<<<1, 64, 0, stream>>>(tpb, histb,
                                     ntn_w1, ntn_b1, ntn_w2, ntn_b2, ntn_w3, ntn_b3,
                                     hist_w1, hist_b1, hist_w2, hist_b2,
                                     fus_w1, fus_b1, fus_w2, fus_b2,
                                     (float*)d_out);
}
